// SVDHead_37194416783730
// MI455X (gfx1250) — matrix operand order repaired and checked by reading, not compiled, not run on hardware
//
#include <hip/hip_runtime.h>
#include <hip/hip_bf16.h>

typedef __attribute__((ext_vector_type(16))) _Float16 v16h;
typedef __attribute__((ext_vector_type(8)))  float    v8f;

#define NPTS 256
#define SENT 256

// ---- dynamic LDS layout (bytes) ----
constexpr int SMH_OFF  = 0;        // f16 [256][256] softmax matrix   (131072)
constexpr int HH_OFF   = 131072;   // f16 [16][256]  tgt*cnt, padded  (8192)
constexpr int SCF_OFF  = 139264;   // f32 [3][256]   src_corr        (3072)
constexpr int VF_OFF   = 142336;   // f32 [256]      valid flags     (1024)
constexpr int CNT_OFF  = 143360;   // i32 [256]      multiplicities  (1024)
constexpr int SRED_OFF = 144384;   // f32 [8][16]    wave partials   (512)
constexpr int FIN_OFF  = 144896;   // f32 [16]       block sums      (64)
constexpr int MC_OFF   = 144960;   // i32            valid count     (4)
constexpr size_t SMEM_BYTES = 145024;

__device__ __forceinline__ float wsum(float x) {
#pragma unroll
  for (int off = 16; off > 0; off >>= 1) x += __shfl_xor(x, off, 32);
  return x;
}
__device__ __forceinline__ float wmaxr(float x) {
#pragma unroll
  for (int off = 16; off > 0; off >>= 1) x = fmaxf(x, __shfl_xor(x, off, 32));
  return x;
}

template <int NV>
__device__ __forceinline__ void blockReduce(const float* v, float* sred, float* fin,
                                            int t, int lane, int w) {
#pragma unroll
  for (int i = 0; i < NV; ++i) {
    float r = wsum(v[i]);
    if (lane == 0) sred[w * 16 + i] = r;
  }
  __syncthreads();
  if (t < NV) {
    float s = 0.f;
#pragma unroll
    for (int ww = 0; ww < 8; ++ww) s += sred[ww * 16 + t];  // fixed order => deterministic
    fin[t] = s;
  }
  __syncthreads();
}

extern __shared__ char smem_raw[];

__global__ void __launch_bounds__(256)
svdhead_kernel(const float* __restrict__ src, const float* __restrict__ tgt,
               const float* __restrict__ scores, const int* __restrict__ idx0,
               float* __restrict__ outR, float* __restrict__ outT,
               int* __restrict__ outM) {
  const int b = blockIdx.x;
  const int t = threadIdx.x;
  const int lane = t & 31, w = t >> 5;

  _Float16* smH = (_Float16*)(smem_raw + SMH_OFF);
  _Float16* hH  = (_Float16*)(smem_raw + HH_OFF);
  float*    scf = (float*)(smem_raw + SCF_OFF);
  float*    vfS = (float*)(smem_raw + VF_OFF);
  int*      cnt = (int*)(smem_raw + CNT_OFF);
  float*    sred= (float*)(smem_raw + SRED_OFF);
  float*    fin = (float*)(smem_raw + FIN_OFF);
  int*      Mc  = (int*)(smem_raw + MC_OFF);

  // ---------- phase 0: matches -> multiplicity counts ----------
  cnt[t] = 0;
  if (t == 0) *Mc = 0;
  __syncthreads();
  {
    int idx = idx0[(size_t)b * NPTS + t];
    int valid = (idx != SENT);
    int m = valid ? idx : 0;
    vfS[t] = valid ? 1.f : 0.f;
    if (valid) { atomicAdd(&cnt[m], 1); atomicAdd(Mc, 1); }
  }
  __syncthreads();

  // hH[c][i] = f16(tgt[c][i] * cnt[i]); rows 3..15 zero (padding for WMMA B)
  {
    float cw = (float)cnt[t];
    const float* tg = tgt + (size_t)b * 3 * NPTS;
#pragma unroll
    for (int c = 0; c < 3; ++c) hH[c * NPTS + t] = (_Float16)(tg[c * NPTS + t] * cw);
#pragma unroll
    for (int c = 3; c < 16; ++c) hH[c * NPTS + t] = (_Float16)0.f;
  }

  // ---------- phase A: row softmax (f32 in, f16 out to LDS) ----------
  // wave w handles rows [w*32, w*32+32); lane holds cols {2*lane+64q, +1}
  const float* sbase = scores + (size_t)b * NPTS * NPTS;
  for (int rr = 0; rr < 32; ++rr) {
    const int j = w * 32 + rr;
    const float* rp = sbase + (size_t)j * NPTS;
    float x[8];
#pragma unroll
    for (int q = 0; q < 4; ++q) {
      float2 v = *(const float2*)(rp + q * 64 + 2 * lane);
      x[2 * q] = v.x; x[2 * q + 1] = v.y;
    }
    if (rr < 31) __builtin_prefetch(rp + NPTS + 2 * lane, 0, 0);  // stream next row
    float mx = x[0];
#pragma unroll
    for (int i = 1; i < 8; ++i) mx = fmaxf(mx, x[i]);
    mx = wmaxr(mx);
    float e[8], s = 0.f;
#pragma unroll
    for (int i = 0; i < 8; ++i) { e[i] = __expf(x[i] - mx); s += e[i]; }
    s = wsum(s);
    const float inv = 1.f / s;
    _Float16* drow = smH + (size_t)j * NPTS;
#pragma unroll
    for (int q = 0; q < 4; ++q) {
      union { unsigned int u; _Float16 h[2]; } pk;
      pk.h[0] = (_Float16)(e[2 * q] * inv);
      pk.h[1] = (_Float16)(e[2 * q + 1] * inv);
      *(unsigned int*)(drow + q * 64 + 2 * lane) = pk.u;
    }
  }
  __syncthreads();

  // ---------- phase B: src_corr^T[256x16] = SM[256x256] @ (tgt*cnt)^T via WMMA ----------
  // wave w does j-tiles {w, w+8}; K=256 in 8 chunks of 32
  {
    const int colc = lane & 15;
    const int row_in_tile = lane & 15;
    const int asel = (lane < 16) ? 0 : 8;    // A: K sub-chunk select per ISA layout
    const int bsel = (lane < 16) ? 0 : 8;    // B: K sub-chunk select (same as A)
#pragma unroll
    for (int tt = 0; tt < 2; ++tt) {
      const int T = w + tt * 8;
      const int row = T * 16 + row_in_tile;
      v8f acc = {};
#pragma unroll
      for (int kc = 0; kc < 8; ++kc) {
        union { uint4 q[2]; v16h v; } A, Bm;
        const int ka = kc * 32 + asel;
        A.q[0] = *(const uint4*)(smH + (size_t)row * NPTS + ka);        // K ka..ka+7
        A.q[1] = *(const uint4*)(smH + (size_t)row * NPTS + ka + 16);   // K ka+16..ka+23
        const int kb = kc * 32 + bsel;
        Bm.q[0] = *(const uint4*)(hH + colc * NPTS + kb);               // K kb..kb+7
        Bm.q[1] = *(const uint4*)(hH + colc * NPTS + kb + 16);          // K kb+16..kb+23
        acc = __builtin_amdgcn_wmma_f32_16x16x32_f16(
            false, A.v, false, Bm.v, (short)0, acc, false, false);
      }
      // D layout: VGPR r, lane n<16 -> [row T*16+r][col n]; lane n>=16 -> [row T*16+8+r][col n-16]
      const int hi = lane >> 4;
      if (colc < 3) {
#pragma unroll
        for (int r = 0; r < 8; ++r)
          scf[colc * NPTS + T * 16 + hi * 8 + r] = acc[r];
      }
    }
  }
  __syncthreads();

  // ---------- phase C: means and 3x3 cross-covariance (deterministic reductions) ----------
  const float* sp = src + (size_t)b * 3 * NPTS;
  const float s0 = sp[t], s1 = sp[NPTS + t], s2 = sp[2 * NPTS + t];
  const float vv = vfS[t];
  const float c0 = scf[t], c1 = scf[NPTS + t], c2 = scf[2 * NPTS + t];
  {
    float p6[6] = { s0 * vv, s1 * vv, s2 * vv, c0 * vv, c1 * vv, c2 * vv };
    blockReduce<6>(p6, sred, fin, t, lane, w);
  }
  const float Mf = (float)(*Mc);
  const float denom = fmaxf(Mf, 1.f);
  const float sm0 = fin[0] / denom, sm1 = fin[1] / denom, sm2 = fin[2] / denom;
  const float cm0 = fin[3] / denom, cm1 = fin[4] / denom, cm2 = fin[5] / denom;
  {
    const float a0 = (s0 - sm0) * vv, a1 = (s1 - sm1) * vv, a2 = (s2 - sm2) * vv;
    const float b0 = (c0 - cm0) * vv, b1 = (c1 - cm1) * vv, b2 = (c2 - cm2) * vv;
    float p9[9] = { a0 * b0, a0 * b1, a0 * b2,
                    a1 * b0, a1 * b1, a1 * b2,
                    a2 * b0, a2 * b1, a2 * b2 };
    blockReduce<9>(p9, sred, fin, t, lane, w);
  }

  // ---------- phase D: 3x3 SVD (one-sided Jacobi), R, t ----------
  if (t == 0) {
    float A[3][3], V[3][3];
#pragma unroll
    for (int i = 0; i < 3; ++i)
#pragma unroll
      for (int j = 0; j < 3; ++j) { A[i][j] = fin[i * 3 + j]; V[i][j] = (i == j) ? 1.f : 0.f; }

    const int ps[3] = {0, 0, 1}, qs[3] = {1, 2, 2};
    for (int sweep = 0; sweep < 12; ++sweep) {
      for (int pp = 0; pp < 3; ++pp) {
        const int p = ps[pp], q = qs[pp];
        float app = 0.f, aqq = 0.f, apq = 0.f;
#pragma unroll
        for (int i = 0; i < 3; ++i) {
          app += A[i][p] * A[i][p]; aqq += A[i][q] * A[i][q]; apq += A[i][p] * A[i][q];
        }
        if (fabsf(apq) < 1e-20f) continue;
        const float zeta = (aqq - app) / (2.f * apq);
        const float tr = copysignf(1.f, zeta) / (fabsf(zeta) + sqrtf(1.f + zeta * zeta));
        const float cc = 1.f / sqrtf(1.f + tr * tr);
        const float ssn = cc * tr;
#pragma unroll
        for (int i = 0; i < 3; ++i) {
          float ap = A[i][p], aq = A[i][q];
          A[i][p] = cc * ap - ssn * aq; A[i][q] = ssn * ap + cc * aq;
          float vp = V[i][p], vq = V[i][q];
          V[i][p] = cc * vp - ssn * vq; V[i][q] = ssn * vp + cc * vq;
        }
      }
    }
    float sig[3], U[3][3];
#pragma unroll
    for (int k = 0; k < 3; ++k) {
      float n2 = A[0][k] * A[0][k] + A[1][k] * A[1][k] + A[2][k] * A[2][k];
      sig[k] = sqrtf(n2);
      float inv = (sig[k] > 1e-12f) ? 1.f / sig[k] : 0.f;
      for (int i = 0; i < 3; ++i) U[i][k] = A[i][k] * inv;
    }
    // sort singular values descending (joint column permutation of U,V)
    auto swapcol = [&](int a, int bcol) {
      float tsw = sig[a]; sig[a] = sig[bcol]; sig[bcol] = tsw;
      for (int i = 0; i < 3; ++i) {
        tsw = U[i][a]; U[i][a] = U[i][bcol]; U[i][bcol] = tsw;
        tsw = V[i][a]; V[i][a] = V[i][bcol]; V[i][bcol] = tsw;
      }
    };
    if (sig[0] < sig[1]) swapcol(0, 1);
    if (sig[1] < sig[2]) swapcol(1, 2);
    if (sig[0] < sig[1]) swapcol(0, 1);

    float R_[3][3];
#pragma unroll
    for (int i = 0; i < 3; ++i)
      for (int j = 0; j < 3; ++j)
        R_[i][j] = V[i][0] * U[j][0] + V[i][1] * U[j][1] + V[i][2] * U[j][2];
    const float det = R_[0][0] * (R_[1][1] * R_[2][2] - R_[1][2] * R_[2][1])
                    - R_[0][1] * (R_[1][0] * R_[2][2] - R_[1][2] * R_[2][0])
                    + R_[0][2] * (R_[1][0] * R_[2][1] - R_[1][1] * R_[2][0]);
    if (det < 0.f) {
#pragma unroll
      for (int i = 0; i < 3; ++i) V[i][2] = -V[i][2];
#pragma unroll
      for (int i = 0; i < 3; ++i)
        for (int j = 0; j < 3; ++j)
          R_[i][j] = V[i][0] * U[j][0] + V[i][1] * U[j][1] + V[i][2] * U[j][2];
    }
    float tv[3];
#pragma unroll
    for (int i = 0; i < 3; ++i)
      tv[i] = -(R_[i][0] * sm0 + R_[i][1] * sm1 + R_[i][2] * sm2);

    const int Mi = *Mc;
    const bool ok = Mi > 20;
    float* Ro = outR + (size_t)b * 9;
#pragma unroll
    for (int i = 0; i < 3; ++i)
      for (int j = 0; j < 3; ++j)
        Ro[i * 3 + j] = ok ? R_[i][j] : ((i == j) ? 1.f : 0.f);
    float* To = outT + (size_t)b * 3;
#pragma unroll
    for (int i = 0; i < 3; ++i) To[i] = ok ? tv[i] : 0.f;
    outM[b] = Mi;
  }
}

extern "C" void kernel_launch(void* const* d_in, const int* in_sizes, int n_in,
                              void* d_out, int out_size, void* d_ws, size_t ws_size,
                              hipStream_t stream) {
  const float* src    = (const float*)d_in[0];
  const float* tgt    = (const float*)d_in[1];
  const float* scores = (const float*)d_in[2];
  const int*   idx0   = (const int*)d_in[3];
  // d_in[4] (indices1) unused by the reference math.
  const int B = in_sizes[3] / NPTS;

  float* out  = (float*)d_out;
  float* outR = out;                         // [B,3,3]
  float* outT = out + (size_t)B * 9;         // [B,3]
  int*   outM = (int*)(out + (size_t)B * 12);// [B] int32 bits

  static bool attr_set = false;
  (void)attr_set;
  hipFuncSetAttribute((const void*)svdhead_kernel,
                      hipFuncAttributeMaxDynamicSharedMemorySize,
                      (int)SMEM_BYTES);

  svdhead_kernel<<<B, 256, SMEM_BYTES, stream>>>(src, tgt, scores, idx0,
                                                 outR, outT, outM);
}
